// EfficientAttention_39926015984296
// MI455X (gfx1250) — matrix operand order repaired and checked by reading, not compiled, not run on hardware
//
#include <hip/hip_runtime.h>
#include <hip/hip_bf16.h>

// Problem constants (match reference)
#define KT   2048   // sequence length
#define KD   2048   // model dim
#define KH   16     // query heads
#define KKV  4      // kv heads
#define KDH  128    // head dim
#define KWIN 512    // sliding window

typedef __attribute__((ext_vector_type(16))) __bf16 bf16x16;
typedef __attribute__((ext_vector_type(8)))  float  f32x8;

union BF16Frag {
    bf16x16 v;
    uint4 q[2];
    unsigned short u[16];
};

__device__ __forceinline__ unsigned short f2bfu(float f) {
    unsigned u = __builtin_bit_cast(unsigned, f);
    u += 0x7FFFu + ((u >> 16) & 1u);   // round-to-nearest-even
    return (unsigned short)(u >> 16);
}
__device__ __forceinline__ __bf16 f2bf(float f) {
    unsigned short h = f2bfu(f);
    return __builtin_bit_cast(__bf16, h);
}

__device__ __forceinline__ f32x8 wmma_bf16(const bf16x16 a, const bf16x16 b, const f32x8 c) {
    // D = A(16x32 bf16) x B(32x16 bf16) + C(16x16 f32)
    return __builtin_amdgcn_wmma_f32_16x16x32_bf16(
        false, a, false, b, (short)0, c, false, false);
}

// ---------------------------------------------------------------------------
// 1) fp32 -> bf16 convert, 8 elements per thread (b128 in/out)
// ---------------------------------------------------------------------------
__global__ __launch_bounds__(256)
void cvt_bf16x8_kernel(const float* __restrict__ in, __bf16* __restrict__ out, int n8) {
    int i = blockIdx.x * 256 + threadIdx.x;
    if (i >= n8) return;
    const float4* src = (const float4*)in + 2 * (size_t)i;
    float4 a = src[0], b = src[1];
    union { unsigned short u[8]; uint4 q; } r;
    r.u[0] = f2bfu(a.x); r.u[1] = f2bfu(a.y); r.u[2] = f2bfu(a.z); r.u[3] = f2bfu(a.w);
    r.u[4] = f2bfu(b.x); r.u[5] = f2bfu(b.y); r.u[6] = f2bfu(b.z); r.u[7] = f2bfu(b.w);
    *((uint4*)out + i) = r.q;
}

// ---------------------------------------------------------------------------
// Shared WMMA GEMM mainloop: 32x64 wave tile of C = A[MxK] * B[NxK]^T,
// both operands row-major with K contiguous. acc[mi][ni] in WMMA C layout.
// ---------------------------------------------------------------------------
__device__ __forceinline__ void gemm_tile_32x64(const __bf16* __restrict__ A,
                                                const __bf16* __restrict__ B,
                                                int K, int m0, int n0,
                                                int lr, int hi,
                                                f32x8 (&acc)[2][4]) {
#pragma unroll
    for (int mi = 0; mi < 2; ++mi)
#pragma unroll
        for (int ni = 0; ni < 4; ++ni)
#pragma unroll
            for (int r = 0; r < 8; ++r) acc[mi][ni][r] = 0.0f;

    for (int kk = 0; kk < K; kk += 32) {
        BF16Frag a[2], b[4];
#pragma unroll
        for (int mi = 0; mi < 2; ++mi) {
            // A fragment 16x32: lane(hi,lr) holds K = hi*8+e and 16+hi*8+e
            const __bf16* p = A + (size_t)(m0 + mi * 16 + lr) * K + kk + hi * 8;
            a[mi].q[0] = *(const uint4*)p;
            a[mi].q[1] = *(const uint4*)(p + 16);
        }
#pragma unroll
        for (int ni = 0; ni < 4; ++ni) {
            // B fragment 32x16: lane(hi,lr) holds column lr, K = hi*8+e and 16+hi*8+e
            const __bf16* p = B + (size_t)(n0 + ni * 16 + lr) * K + kk + hi * 8;
            b[ni].q[0] = *(const uint4*)p;
            b[ni].q[1] = *(const uint4*)(p + 16);
        }
#pragma unroll
        for (int mi = 0; mi < 2; ++mi)
#pragma unroll
            for (int ni = 0; ni < 4; ++ni)
                acc[mi][ni] = wmma_bf16(a[mi].v, b[ni].v, acc[mi][ni]);
    }
}

// ---------------------------------------------------------------------------
// 2a) GEMM with plain f32 epilogue (used for the final O-projection)
// ---------------------------------------------------------------------------
__global__ __launch_bounds__(128)
void gemm_bf16_kernel(const __bf16* __restrict__ A, const __bf16* __restrict__ B,
                      float* __restrict__ C, int M, int N, int K) {
    const int lane = threadIdx.x & 31;
    const int wave = threadIdx.x >> 5;
    const int hi = lane >> 4, lr = lane & 15;
    const int m0 = blockIdx.y * 64 + (wave >> 1) * 32;
    const int n0 = blockIdx.x * 128 + (wave & 1) * 64;

    f32x8 acc[2][4];
    gemm_tile_32x64(A, B, K, m0, n0, lr, hi, acc);

#pragma unroll
    for (int mi = 0; mi < 2; ++mi)
#pragma unroll
        for (int ni = 0; ni < 4; ++ni) {
            float* cp = C + (size_t)(m0 + mi * 16 + hi * 8) * N + n0 + ni * 16 + lr;
#pragma unroll
            for (int r = 0; r < 8; ++r) cp[(size_t)r * N] = acc[mi][ni][r];
        }
}

// ---------------------------------------------------------------------------
// 2b) Projection GEMM with fused RoPE epilogue:
//     acc = x @ W^T; rope pairs (2i,2i+1) are adjacent lanes in the C tile,
//     so one shfl_xor(.,1) provides the partner. Writes bf16 [heads][T][128].
// ---------------------------------------------------------------------------
__global__ __launch_bounds__(128)
void gemm_rope_kernel(const __bf16* __restrict__ A, const __bf16* __restrict__ B,
                      const float* __restrict__ fc, __bf16* __restrict__ dst,
                      int M, int N, int K) {
    const int lane = threadIdx.x & 31;
    const int wave = threadIdx.x >> 5;
    const int hi = lane >> 4, lr = lane & 15;
    const int m0 = blockIdx.y * 64 + (wave >> 1) * 32;
    const int n0 = blockIdx.x * 128 + (wave & 1) * 64;

    f32x8 acc[2][4];
    gemm_tile_32x64(A, B, K, m0, n0, lr, hi, acc);

#pragma unroll
    for (int mi = 0; mi < 2; ++mi)
#pragma unroll
        for (int ni = 0; ni < 4; ++ni) {
            int c = n0 + ni * 16 + lr;      // global output column
            int d = c & (KDH - 1);          // dim within head
            int h = c >> 7;                 // head
            int pi = d >> 1;                // rope pair index
            bool even = (d & 1) == 0;
#pragma unroll
            for (int r = 0; r < 8; ++r) {
                int t = m0 + mi * 16 + hi * 8 + r;
                float cs = fc[(t * 64 + pi) * 2 + 0];
                float sn = fc[(t * 64 + pi) * 2 + 1];
                float v  = acc[mi][ni][r];
                float pv = __shfl_xor(v, 1, 32);   // partner column (c ^ 1)
                // even d: o1 = x1*cos - x2*sin ; odd d: o2 = x1*sin + x2*cos
                float o = even ? (v * cs - pv * sn) : (pv * sn + v * cs);
                dst[((size_t)h * KT + t) * KDH + d] = f2bf(o);
            }
        }
}

// ---------------------------------------------------------------------------
// 2c) V projection GEMM with fused transpose epilogue -> Vt [KV*128][T] bf16
// ---------------------------------------------------------------------------
__global__ __launch_bounds__(128)
void gemm_vtrans_kernel(const __bf16* __restrict__ A, const __bf16* __restrict__ B,
                        __bf16* __restrict__ dst, int M, int N, int K) {
    const int lane = threadIdx.x & 31;
    const int wave = threadIdx.x >> 5;
    const int hi = lane >> 4, lr = lane & 15;
    const int m0 = blockIdx.y * 64 + (wave >> 1) * 32;
    const int n0 = blockIdx.x * 128 + (wave & 1) * 64;

    f32x8 acc[2][4];
    gemm_tile_32x64(A, B, K, m0, n0, lr, hi, acc);

#pragma unroll
    for (int mi = 0; mi < 2; ++mi)
#pragma unroll
        for (int ni = 0; ni < 4; ++ni) {
            int c = n0 + ni * 16 + lr;      // = g*128 + d
#pragma unroll
            for (int r = 0; r < 8; ++r) {
                int t = m0 + mi * 16 + hi * 8 + r;
                dst[(size_t)c * KT + t] = f2bf(acc[mi][ni][r]);
            }
        }
}

// ---------------------------------------------------------------------------
// 3) Sliding-window flash attention, 1 wave per (head, 16-query tile).
//    Q  [H][T][128]  bf16 (rope'd)   Kc [KV][T][128] bf16 (rope'd)
//    Vt [KV][128][T] bf16            Y  [T][2048]    bf16
// ---------------------------------------------------------------------------
__global__ __launch_bounds__(32)
void attn_kernel(const __bf16* __restrict__ Q, const __bf16* __restrict__ Kc,
                 const __bf16* __restrict__ Vt, __bf16* __restrict__ Y) {
    __shared__ unsigned short pTile[32 * 16];   // column-major: [key k][query m]

    const int lane = threadIdx.x;
    const int hi = lane >> 4;
    const int lr = lane & 15;
    const int t0 = blockIdx.x * 16;             // query tile base
    const int h  = blockIdx.y;                  // query head
    const int g  = h / (KH / KKV);              // kv head

    const __bf16* Qh = Q  + (size_t)h * KT * KDH;
    const __bf16* Kg = Kc + (size_t)g * KT * KDH;
    const __bf16* Vg = Vt + (size_t)g * KDH * KT;

    // Q tile 16x128 as 4 A-fragments (K=32 each)
    BF16Frag qa[4];
#pragma unroll
    for (int kd = 0; kd < 4; ++kd) {
        const __bf16* p = Qh + (size_t)(t0 + lr) * KDH + kd * 32 + hi * 8;
        qa[kd].q[0] = *(const uint4*)p;
        qa[kd].q[1] = *(const uint4*)(p + 16);
    }

    f32x8 o[8];
#pragma unroll
    for (int nd = 0; nd < 8; ++nd)
#pragma unroll
        for (int r = 0; r < 8; ++r) o[nd][r] = 0.0f;

    float mrow[8], lrow[8];
#pragma unroll
    for (int r = 0; r < 8; ++r) { mrow[r] = -1e30f; lrow[r] = 0.0f; }

    const float scale = 0.08838834764831845f;   // 1/sqrt(128)

    int jstart = t0 - (KWIN - 1);
    if (jstart < 0) jstart = 0;
    jstart &= ~31;                               // align to key-block

    for (int jb = jstart; jb <= t0 + 15; jb += 32) {
        // ---- S = Q K^T for a 16x32 score tile (two 16x16 fragments) ----
        f32x8 sfr[2];
#pragma unroll
        for (int f = 0; f < 2; ++f) {
#pragma unroll
            for (int r = 0; r < 8; ++r) sfr[f][r] = 0.0f;
            int krow = jb + f * 16 + lr;
            if (krow > KT - 1) krow = KT - 1;    // clamp (masked anyway)
#pragma unroll
            for (int kd = 0; kd < 4; ++kd) {
                BF16Frag b;
                const __bf16* p = Kg + (size_t)krow * KDH + kd * 32 + hi * 8;
                b.q[0] = *(const uint4*)p;
                b.q[1] = *(const uint4*)(p + 16);
                sfr[f] = wmma_bf16(qa[kd].v, b.v, sfr[f]);
            }
        }

        // ---- online softmax: new row max (rows r+8*hi, columns across lanes) ----
        float mnew[8];
#pragma unroll
        for (int r = 0; r < 8; ++r) {
            int i = t0 + 8 * hi + r;
            float mx = -1e30f;
#pragma unroll
            for (int f = 0; f < 2; ++f) {
                int j = jb + f * 16 + lr;
                bool ok = (j <= i) && (j > i - KWIN);
                float sv = ok ? sfr[f][r] * scale : -1e30f;
                sfr[f][r] = sv;
                mx = fmaxf(mx, sv);
            }
#pragma unroll
            for (int off = 1; off < 16; off <<= 1)
                mx = fmaxf(mx, __shfl_xor(mx, off, 32));
            mnew[r] = fmaxf(mrow[r], mx);
        }

        // rescale running accumulators
#pragma unroll
        for (int r = 0; r < 8; ++r) {
            float corr = __expf(mrow[r] - mnew[r]);
            lrow[r] *= corr;
#pragma unroll
            for (int nd = 0; nd < 8; ++nd) o[nd][r] *= corr;
            mrow[r] = mnew[r];
        }

        // ---- P = exp(S - m), masked entries exactly 0; stash in LDS (col-major) ----
        float psum[8];
#pragma unroll
        for (int r = 0; r < 8; ++r) psum[r] = 0.0f;
#pragma unroll
        for (int f = 0; f < 2; ++f) {
            int j = jb + f * 16 + lr;
            BF16Frag pk;
#pragma unroll
            for (int r = 0; r < 8; ++r) {
                int i = t0 + 8 * hi + r;
                bool ok = (j <= i) && (j > i - KWIN);
                float p = ok ? __expf(sfr[f][r] - mnew[r]) : 0.0f;
                psum[r] += p;
                pk.u[r] = f2bfu(p);
            }
            // column f*16+lr, rows 8*hi..8*hi+7  (16-byte aligned store)
            *(uint4*)&pTile[(f * 16 + lr) * 16 + hi * 8] = pk.q[0];
        }
#pragma unroll
        for (int r = 0; r < 8; ++r) {
            float s = psum[r];
#pragma unroll
            for (int off = 1; off < 16; off <<= 1)
                s += __shfl_xor(s, off, 32);
            lrow[r] += s;
        }
        __syncthreads();

        // ---- reload P as A-fragment (16x32) from LDS transpose ----
        BF16Frag pa;
#pragma unroll
        for (int e = 0; e < 8; ++e) {
            pa.u[e]     = pTile[(hi * 8 + e) * 16 + lr];       // K = hi*8+e
            pa.u[e + 8] = pTile[(16 + hi * 8 + e) * 16 + lr];  // K = 16+hi*8+e
        }
        __syncthreads();

        // ---- O += P * V  (8 DH-chunks of 16) ----
#pragma unroll
        for (int nd = 0; nd < 8; ++nd) {
            BF16Frag vb;
            const __bf16* p = Vg + (size_t)(nd * 16 + lr) * KT + jb + hi * 8;
            vb.q[0] = *(const uint4*)p;      // keys beyond T read pad; P=0 there
            vb.q[1] = *(const uint4*)(p + 16);
            o[nd] = wmma_bf16(pa.v, vb.v, o[nd]);
        }
    }

    // ---- epilogue: O / l -> Y[t][h*128 + d] bf16 ----
#pragma unroll
    for (int r = 0; r < 8; ++r) {
        int i = t0 + 8 * hi + r;
        float inv = 1.0f / lrow[r];
#pragma unroll
        for (int nd = 0; nd < 8; ++nd)
            Y[(size_t)i * KD + h * KDH + nd * 16 + lr] = f2bf(o[nd][r] * inv);
    }
}

// ---------------------------------------------------------------------------
// Host launch
// ---------------------------------------------------------------------------
static inline char* carve(char*& p, size_t bytes) {
    char* r = p;
    p += (bytes + 255) & ~(size_t)255;
    return r;
}

extern "C" void kernel_launch(void* const* d_in, const int* in_sizes, int n_in,
                              void* d_out, int out_size, void* d_ws, size_t ws_size,
                              hipStream_t stream) {
    const float* x  = (const float*)d_in[0];   // [1, 2048, 2048]
    const float* fc = (const float*)d_in[1];   // [2048, 64, 2]
    const float* Wq = (const float*)d_in[2];   // [2048, 2048]
    const float* Wk = (const float*)d_in[3];   // [512, 2048]
    const float* Wv = (const float*)d_in[4];   // [512, 2048]
    const float* Wo = (const float*)d_in[5];   // [2048, 2048]
    float* out = (float*)d_out;                // [1, 2048, 2048]

    char* p = (char*)d_ws;
    __bf16* Xb  = (__bf16*)carve(p, (size_t)KT * KD * 2);
    __bf16* Wqb = (__bf16*)carve(p, (size_t)KD * KD * 2);
    __bf16* Wkb = (__bf16*)carve(p, (size_t)KKV * KDH * KD * 2);
    __bf16* Wvb = (__bf16*)carve(p, (size_t)KKV * KDH * KD * 2);
    __bf16* Wob = (__bf16*)carve(p, (size_t)KD * KD * 2);
    __bf16* Qr  = (__bf16*)carve(p, (size_t)KH * KT * KDH * 2);
    __bf16* Kr  = (__bf16*)carve(p, (size_t)KKV * KT * KDH * 2);
    __bf16* Vt  = (__bf16*)carve(p, (size_t)KKV * KDH * KT * 2 + 4096); // +pad for OOB key blocks
    __bf16* Yb  = (__bf16*)carve(p, (size_t)KT * KD * 2);

    const int CB = 256;
    auto blocks8 = [](int n) { return (n / 8 + 255) / 256; };

    // 1) fp32 -> bf16 converts (8 elems/thread)
    cvt_bf16x8_kernel<<<blocks8(KT * KD), CB, 0, stream>>>(x,  Xb,  KT * KD / 8);
    cvt_bf16x8_kernel<<<blocks8(KD * KD), CB, 0, stream>>>(Wq, Wqb, KD * KD / 8);
    cvt_bf16x8_kernel<<<blocks8(KKV * KDH * KD), CB, 0, stream>>>(Wk, Wkb, KKV * KDH * KD / 8);
    cvt_bf16x8_kernel<<<blocks8(KKV * KDH * KD), CB, 0, stream>>>(Wv, Wvb, KKV * KDH * KD / 8);
    cvt_bf16x8_kernel<<<blocks8(KD * KD), CB, 0, stream>>>(Wo, Wob, KD * KD / 8);

    // 2) projections with fused RoPE / transpose epilogues (no f32 round trip)
    gemm_rope_kernel<<<dim3(KD / 128, KT / 64), 128, 0, stream>>>(Xb, Wqb, fc, Qr, KT, KD, KD);
    gemm_rope_kernel<<<dim3((KKV * KDH) / 128, KT / 64), 128, 0, stream>>>(Xb, Wkb, fc, Kr, KT, KKV * KDH, KD);
    gemm_vtrans_kernel<<<dim3((KKV * KDH) / 128, KT / 64), 128, 0, stream>>>(Xb, Wvb, Vt, KT, KKV * KDH, KD);

    // 3) sliding-window attention -> Yb [T][2048] bf16
    attn_kernel<<<dim3(KT / 16, KH), 32, 0, stream>>>(Qr, Kr, Vt, Yb);

    // 4) output projection: out = Yb @ Wo^T (f32)
    gemm_bf16_kernel<<<dim3(KD / 128, KT / 64), 128, 0, stream>>>(Yb, Wob, out, KT, KD, KD);
}
